// TransformerModel_59107339927885
// MI455X (gfx1250) — matrix operand order repaired and checked by reading, not compiled, not run on hardware
//
#include <hip/hip_runtime.h>
#include <math.h>

typedef __bf16 bf16;
typedef __attribute__((ext_vector_type(16))) __bf16 v16bf;
typedef __attribute__((ext_vector_type(8)))  __bf16 bf16x8;
typedef __attribute__((ext_vector_type(8)))  float  v8f;
typedef __attribute__((ext_vector_type(4)))  unsigned int u32x4;

#define S_LEN 448
#define BATCH 64
#define FEAT 8
#define DMODEL 256
#define NHEAD 8
#define DKH 32
#define NLAYER 4
#define NTOK (S_LEN*BATCH)
#define PREDN 96
#define LNEPS 1e-5f

// ---------------------------------------------------------------------------
// CDNA5 async global->LDS copy (ASYNCcnt path), with sync fallback.
// Builtin expects pointers to 16-byte int vectors (per hipcc diagnostic).
// ---------------------------------------------------------------------------
#if defined(__gfx1250__) && defined(__has_builtin)
#if __has_builtin(__builtin_amdgcn_global_load_async_to_lds_b128) && \
    __has_builtin(__builtin_amdgcn_s_wait_asynccnt)
#define USE_ASYNC_LDS 1
#endif
#endif
#ifndef USE_ASYNC_LDS
#define USE_ASYNC_LDS 0
#endif

typedef int i32x4v __attribute__((__vector_size__(4 * sizeof(int))));

static __device__ __forceinline__ void copy16_to_lds(const bf16* g, bf16* l) {
#if USE_ASYNC_LDS
  __builtin_amdgcn_global_load_async_to_lds_b128(
      (__attribute__((address_space(1))) i32x4v*)(const void*)g,
      (__attribute__((address_space(3))) i32x4v*)(void*)l, 0, 0);
#else
  *(u32x4*)l = *(const u32x4*)g;
#endif
}
static __device__ __forceinline__ void async_wait() {
#if USE_ASYNC_LDS
  __builtin_amdgcn_s_wait_asynccnt(0);
#endif
}

// ---------------------------------------------------------------------------
// WMMA fragment helpers (CDNA5 wave32 layouts, 16-bit A 16x32 / B 32x16)
// ---------------------------------------------------------------------------
static __device__ __forceinline__ v16bf cat8(bf16x8 lo, bf16x8 hi) {
  v16bf r;
#pragma unroll
  for (int i = 0; i < 8; ++i) { r[i] = lo[i]; r[i + 8] = hi[i]; }
  return r;
}
static __device__ __forceinline__ v16bf frag_ld(const bf16* p0, const bf16* p1) {
  bf16x8 lo = *(const bf16x8*)p0;
  bf16x8 hi = *(const bf16x8*)p1;
  return cat8(lo, hi);
}

// ---------------------------------------------------------------------------
// fp32 -> bf16 weight conversion
// ---------------------------------------------------------------------------
__global__ __launch_bounds__(256) void cvt_kernel(const float* __restrict__ in,
                                                  bf16* __restrict__ out, int n) {
  int i = blockIdx.x * 256 + threadIdx.x;
  if (i < n) out[i] = (bf16)in[i];
}

// ---------------------------------------------------------------------------
// x = src @ w_in + b_in + pos_enc  -> X (f32) and XB (bf16)
// ---------------------------------------------------------------------------
__global__ __launch_bounds__(256) void input_pe_kernel(
    const float* __restrict__ src, const float* __restrict__ w_in,
    const float* __restrict__ b_in, float* __restrict__ X, bf16* __restrict__ XB) {
  int n = blockIdx.x;       // token = s*BATCH + b
  int d = threadIdx.x;      // model dim
  int s = n / BATCH;
  float acc = b_in[d];
#pragma unroll
  for (int f = 0; f < FEAT; ++f) acc += src[n * FEAT + f] * w_in[f * DMODEL + d];
  float divv = __expf(-(float)(d & ~1) * (9.210340371976184f / (float)DMODEL));
  float ang = (float)s * divv;
  acc += (d & 1) ? __cosf(ang) : __sinf(ang);
  X[(size_t)n * DMODEL + d] = acc;
  XB[(size_t)n * DMODEL + d] = (bf16)acc;
}

// ---------------------------------------------------------------------------
// WMMA GEMM: Y[N,256] = Xb[N,256] @ Wb[256,256] + bias, opt relu,
// output fp32 (Yf) and/or bf16 (Yb).
// Block tile 256x64, 8 waves, each wave computes 32x64 (8 accumulators).
// A staged row-major (k-contig) via async-to-LDS; B staged transposed [n][k].
// ---------------------------------------------------------------------------
__global__ __launch_bounds__(256) void gemm_bf16_kernel(
    const bf16* __restrict__ X, const bf16* __restrict__ W,
    const float* __restrict__ bias, float* __restrict__ Yf,
    bf16* __restrict__ Yb, int relu) {
  __shared__ __align__(16) bf16 As[256 * 40];  // [row][k]
  __shared__ __align__(16) bf16 Bs[64 * 40];   // [n][k] (transposed)
  const int t = threadIdx.x;
  const int wid = t >> 5, lane = t & 31;
  const int half = lane >> 4, ln = lane & 15;
  const int row0 = blockIdx.x * 256, col0 = blockIdx.y * 64;

  v8f acc[2][4];
#pragma unroll
  for (int i = 0; i < 2; ++i)
#pragma unroll
    for (int j = 0; j < 4; ++j) acc[i][j] = (v8f){0, 0, 0, 0, 0, 0, 0, 0};

  const int bk = t & 31, bn = (t >> 5) * 8;  // B staging: 32 k-rows x 8 n-chunks of 8

  for (int k0 = 0; k0 < DMODEL; k0 += 32) {
    // A tile 256x32: one row per thread, 4x 16B async chunks
    const bf16* ga = &X[(size_t)(row0 + t) * DMODEL + k0];
#pragma unroll
    for (int c = 0; c < 32; c += 8) copy16_to_lds(ga + c, &As[t * 40 + c]);
    // B tile 32x64 transposed into [n][k]
    bf16x8 wv = *(const bf16x8*)&W[(size_t)(k0 + bk) * DMODEL + col0 + bn];
#pragma unroll
    for (int i = 0; i < 8; ++i) Bs[(bn + i) * 40 + bk] = wv[i];
    async_wait();
    __syncthreads();

    const bf16* pa0 = &As[(wid * 32 + ln) * 40 + half * 8];
    v16bf a0 = frag_ld(pa0, pa0 + 16);
    const bf16* pa1 = &As[(wid * 32 + 16 + ln) * 40 + half * 8];
    v16bf a1 = frag_ld(pa1, pa1 + 16);
#pragma unroll
    for (int j = 0; j < 4; ++j) {
      const bf16* pb = &Bs[(j * 16 + ln) * 40 + half * 8];
      v16bf b = frag_ld(pb, pb + 16);
      acc[0][j] = __builtin_amdgcn_wmma_f32_16x16x32_bf16(false, a0, false, b, (short)0, acc[0][j], false, false);
      acc[1][j] = __builtin_amdgcn_wmma_f32_16x16x32_bf16(false, a1, false, b, (short)0, acc[1][j], false, false);
    }
    __syncthreads();
  }

#pragma unroll
  for (int j = 0; j < 4; ++j) {
    int gc = col0 + j * 16 + ln;
    float bv = bias[gc];
#pragma unroll
    for (int i = 0; i < 2; ++i) {
#pragma unroll
      for (int r = 0; r < 8; ++r) {
        int grow = row0 + wid * 32 + i * 16 + r + 8 * half;
        float v = acc[i][j][r] + bv;
        if (relu) v = fmaxf(v, 0.f);
        size_t o = (size_t)grow * DMODEL + gc;
        if (Yf) Yf[o] = v;
        if (Yb) Yb[o] = (bf16)v;
      }
    }
  }
}

// ---------------------------------------------------------------------------
// Flash attention: block = (b,h) x 64 query rows; 4 waves x 16 rows each.
// Q/K/V bf16 in [S,B,256] layout (head h = cols h*32..h*32+31).
// ---------------------------------------------------------------------------
__global__ __launch_bounds__(128) void attn_kernel(
    const bf16* __restrict__ Q, const bf16* __restrict__ K,
    const bf16* __restrict__ V, bf16* __restrict__ CTX) {
  __shared__ __align__(16) bf16 Qt[64 * 40];      // [q][d]
  __shared__ __align__(16) bf16 Kt[32 * 40];      // [key][d]
  __shared__ __align__(16) bf16 Vt[32 * 40];      // [d][key] (transposed)
  __shared__ __align__(16) bf16 Pt[4 * 16 * 40];  // per-wave P scratch [q][key]

  const int t = threadIdx.x;
  const int wid = t >> 5, lane = t & 31;
  const int half = lane >> 4, ln = lane & 15;
  const int bh = blockIdx.x;
  const int b = bh >> 3, h = bh & 7;
  const int q0 = blockIdx.y * 64;
  const float scale = 0.1767766952966369f;  // 1/sqrt(32)

  {  // stage Q tile (64 x 32) via async copies
    int row = t >> 1, c = (t & 1) * 16;
    const bf16* g = &Q[((size_t)(q0 + row) * BATCH + b) * DMODEL + h * DKH + c];
    copy16_to_lds(g, &Qt[row * 40 + c]);
    copy16_to_lds(g + 8, &Qt[row * 40 + c + 8]);
  }
  async_wait();
  __syncthreads();

  const bf16* pq = &Qt[(wid * 16 + ln) * 40 + half * 8];
  v16bf aq = frag_ld(pq, pq + 16);

  v8f acc0 = {0, 0, 0, 0, 0, 0, 0, 0}, acc1 = {0, 0, 0, 0, 0, 0, 0, 0};
  float mst[8], lst[8];
#pragma unroll
  for (int r = 0; r < 8; ++r) { mst[r] = -1e30f; lst[r] = 0.f; }

  bf16* Pw = &Pt[wid * 640];
  const int kr = t >> 2, kc = (t & 3) * 8;

  for (int kb = 0; kb < S_LEN; kb += 32) {
    // stage K row-major (async), V transposed (manual scatter)
    copy16_to_lds(&K[((size_t)(kb + kr) * BATCH + b) * DMODEL + h * DKH + kc],
                  &Kt[kr * 40 + kc]);
    bf16x8 vv = *(const bf16x8*)&V[((size_t)(kb + kr) * BATCH + b) * DMODEL + h * DKH + kc];
#pragma unroll
    for (int i = 0; i < 8; ++i) Vt[(kc + i) * 40 + kr] = vv[i];
    async_wait();
    __syncthreads();

    const bf16* pk0 = &Kt[ln * 40 + half * 8];
    v16bf bk0 = frag_ld(pk0, pk0 + 16);
    const bf16* pk1 = &Kt[(16 + ln) * 40 + half * 8];
    v16bf bk1 = frag_ld(pk1, pk1 + 16);
    v8f z = {0, 0, 0, 0, 0, 0, 0, 0};
    v8f s0 = __builtin_amdgcn_wmma_f32_16x16x32_bf16(false, aq, false, bk0, (short)0, z, false, false);
    v8f s1 = __builtin_amdgcn_wmma_f32_16x16x32_bf16(false, aq, false, bk1, (short)0, z, false, false);

#pragma unroll
    for (int r = 0; r < 8; ++r) {
      float e0 = s0[r] * scale, e1 = s1[r] * scale;
      float mx = fmaxf(e0, e1);
      mx = fmaxf(mx, __shfl_xor(mx, 1));
      mx = fmaxf(mx, __shfl_xor(mx, 2));
      mx = fmaxf(mx, __shfl_xor(mx, 4));
      mx = fmaxf(mx, __shfl_xor(mx, 8));
      float mnew = fmaxf(mst[r], mx);
      float al = __expf(mst[r] - mnew);
      float p0 = __expf(e0 - mnew);
      float p1 = __expf(e1 - mnew);
      float ps = p0 + p1;
      ps += __shfl_xor(ps, 1);
      ps += __shfl_xor(ps, 2);
      ps += __shfl_xor(ps, 4);
      ps += __shfl_xor(ps, 8);
      lst[r] = lst[r] * al + ps;
      mst[r] = mnew;
      acc0[r] *= al;
      acc1[r] *= al;
      Pw[(r + 8 * half) * 40 + ln] = (bf16)p0;
      Pw[(r + 8 * half) * 40 + 16 + ln] = (bf16)p1;
    }

    const bf16* pp = &Pw[ln * 40 + half * 8];
    v16bf ap = frag_ld(pp, pp + 16);
    const bf16* pv0 = &Vt[ln * 40 + half * 8];
    v16bf bv0 = frag_ld(pv0, pv0 + 16);
    const bf16* pv1 = &Vt[(16 + ln) * 40 + half * 8];
    v16bf bv1 = frag_ld(pv1, pv1 + 16);
    acc0 = __builtin_amdgcn_wmma_f32_16x16x32_bf16(false, ap, false, bv0, (short)0, acc0, false, false);
    acc1 = __builtin_amdgcn_wmma_f32_16x16x32_bf16(false, ap, false, bv1, (short)0, acc1, false, false);
    __syncthreads();
  }

#pragma unroll
  for (int r = 0; r < 8; ++r) {
    float inv = 1.0f / lst[r];
    int gq = q0 + wid * 16 + r + 8 * half;
    size_t base = ((size_t)gq * BATCH + b) * DMODEL + h * DKH;
    CTX[base + ln] = (bf16)(acc0[r] * inv);
    CTX[base + 16 + ln] = (bf16)(acc1[r] * inv);
  }
}

// ---------------------------------------------------------------------------
// x = LayerNorm(Xin + T) * g + be   -> Xout (f32) and Xb (bf16).  Wave per row.
// ---------------------------------------------------------------------------
__global__ __launch_bounds__(256) void add_ln_kernel(
    const float* __restrict__ Xin, const float* __restrict__ T,
    const float* __restrict__ g, const float* __restrict__ be,
    float* __restrict__ Xout, bf16* __restrict__ Xb) {
  int wid = threadIdx.x >> 5, lane = threadIdx.x & 31;
  int row = blockIdx.x * 8 + wid;
  float v[8];
  float s = 0.f;
#pragma unroll
  for (int i = 0; i < 8; ++i) {
    int d = lane + i * 32;
    v[i] = Xin[(size_t)row * DMODEL + d] + T[(size_t)row * DMODEL + d];
    s += v[i];
  }
  s += __shfl_xor(s, 1); s += __shfl_xor(s, 2); s += __shfl_xor(s, 4);
  s += __shfl_xor(s, 8); s += __shfl_xor(s, 16);
  float mu = s * (1.0f / DMODEL);
  float q = 0.f;
#pragma unroll
  for (int i = 0; i < 8; ++i) { float dd = v[i] - mu; q += dd * dd; }
  q += __shfl_xor(q, 1); q += __shfl_xor(q, 2); q += __shfl_xor(q, 4);
  q += __shfl_xor(q, 8); q += __shfl_xor(q, 16);
  float rstd = rsqrtf(q * (1.0f / DMODEL) + LNEPS);
#pragma unroll
  for (int i = 0; i < 8; ++i) {
    int d = lane + i * 32;
    float y = (v[i] - mu) * rstd * g[d] + be[d];
    Xout[(size_t)row * DMODEL + d] = y;
    Xb[(size_t)row * DMODEL + d] = (bf16)y;
  }
}

// ---------------------------------------------------------------------------
// out[p,b,f] = X[(S-96+p)*B+b] . w_out[:,f] + b_out[f]
// ---------------------------------------------------------------------------
__global__ __launch_bounds__(256) void out_proj_kernel(
    const float* __restrict__ X, const float* __restrict__ w_out,
    const float* __restrict__ b_out, float* __restrict__ out) {
  int tid = blockIdx.x * 256 + threadIdx.x;  // p*512 + b*8 + f
  int f = tid & 7;
  int b = (tid >> 3) & 63;
  int p = tid >> 9;
  int n = (S_LEN - PREDN + p) * BATCH + b;
  float acc = b_out[f];
  for (int d = 0; d < DMODEL; ++d)
    acc += X[(size_t)n * DMODEL + d] * w_out[d * FEAT + f];
  out[tid] = acc;
}

// ---------------------------------------------------------------------------
extern "C" void kernel_launch(void* const* d_in, const int* in_sizes, int n_in,
                              void* d_out, int out_size, void* d_ws, size_t ws_size,
                              hipStream_t stream) {
  (void)in_sizes; (void)n_in; (void)out_size; (void)ws_size;
  const float* src  = (const float*)d_in[0];
  const float* w_in = (const float*)d_in[1];
  const float* b_in = (const float*)d_in[2];
  const float* Wq = (const float*)d_in[3];
  const float* bq = (const float*)d_in[4];
  const float* Wk = (const float*)d_in[5];
  const float* bk = (const float*)d_in[6];
  const float* Wv = (const float*)d_in[7];
  const float* bv = (const float*)d_in[8];
  const float* Wo = (const float*)d_in[9];
  const float* bo = (const float*)d_in[10];
  const float* g1 = (const float*)d_in[11];
  const float* be1 = (const float*)d_in[12];
  const float* W1 = (const float*)d_in[13];
  const float* b1 = (const float*)d_in[14];
  const float* W2 = (const float*)d_in[15];
  const float* b2 = (const float*)d_in[16];
  const float* g2 = (const float*)d_in[17];
  const float* be2 = (const float*)d_in[18];
  const float* w_out = (const float*)d_in[19];
  const float* b_out = (const float*)d_in[20];

  char* ws = (char*)d_ws;
  const size_t F32B = (size_t)NTOK * DMODEL * 4;   // 29,360,128 B
  const size_t BF16B = (size_t)NTOK * DMODEL * 2;  // 14,680,064 B
  float* X = (float*)(ws);
  bf16* XB = (bf16*)(ws + F32B);
  bf16* QB = (bf16*)(ws + F32B + 1 * BF16B);
  bf16* KB = (bf16*)(ws + F32B + 2 * BF16B);
  bf16* VB = (bf16*)(ws + F32B + 3 * BF16B);
  bf16* CB = (bf16*)(ws + F32B + 4 * BF16B);
  bf16* WB = (bf16*)(ws + F32B + 5 * BF16B);
  float* T  = (float*)(ws + F32B + 1 * BF16B);     // aliases QB+KB (dead by then)
  bf16* FFB = (bf16*)(ws + F32B + 3 * BF16B);      // aliases VB (dead by then)

  const int WN = NLAYER * DMODEL * DMODEL;  // 262144 per weight family
  bf16* WBq = WB;
  bf16* WBk = WB + (size_t)WN;
  bf16* WBv = WB + (size_t)2 * WN;
  bf16* WBo = WB + (size_t)3 * WN;
  bf16* WB1 = WB + (size_t)4 * WN;
  bf16* WB2 = WB + (size_t)5 * WN;

  cvt_kernel<<<WN / 256, 256, 0, stream>>>(Wq, WBq, WN);
  cvt_kernel<<<WN / 256, 256, 0, stream>>>(Wk, WBk, WN);
  cvt_kernel<<<WN / 256, 256, 0, stream>>>(Wv, WBv, WN);
  cvt_kernel<<<WN / 256, 256, 0, stream>>>(Wo, WBo, WN);
  cvt_kernel<<<WN / 256, 256, 0, stream>>>(W1, WB1, WN);
  cvt_kernel<<<WN / 256, 256, 0, stream>>>(W2, WB2, WN);

  input_pe_kernel<<<NTOK, 256, 0, stream>>>(src, w_in, b_in, X, XB);

  dim3 ggrid(NTOK / 256, DMODEL / 64);
  for (int l = 0; l < NLAYER; ++l) {
    size_t wo_ = (size_t)l * DMODEL * DMODEL;
    gemm_bf16_kernel<<<ggrid, 256, 0, stream>>>(XB, WBq + wo_, bq + l * DMODEL, nullptr, QB, 0);
    gemm_bf16_kernel<<<ggrid, 256, 0, stream>>>(XB, WBk + wo_, bk + l * DMODEL, nullptr, KB, 0);
    gemm_bf16_kernel<<<ggrid, 256, 0, stream>>>(XB, WBv + wo_, bv + l * DMODEL, nullptr, VB, 0);
    attn_kernel<<<dim3(BATCH * NHEAD, S_LEN / 64), 128, 0, stream>>>(QB, KB, VB, CB);
    gemm_bf16_kernel<<<ggrid, 256, 0, stream>>>(CB, WBo + wo_, bo + l * DMODEL, T, nullptr, 0);
    add_ln_kernel<<<NTOK / 8, 256, 0, stream>>>(X, T, g1 + l * DMODEL, be1 + l * DMODEL, X, XB);
    gemm_bf16_kernel<<<ggrid, 256, 0, stream>>>(XB, WB1 + wo_, b1 + l * DMODEL, nullptr, FFB, 1);
    gemm_bf16_kernel<<<ggrid, 256, 0, stream>>>(FFB, WB2 + wo_, b2 + l * DMODEL, T, nullptr, 0);
    add_ln_kernel<<<NTOK / 8, 256, 0, stream>>>(X, T, g2 + l * DMODEL, be2 + l * DMODEL, X, XB);
  }

  out_proj_kernel<<<(PREDN * BATCH * FEAT) / 256, 256, 0, stream>>>(X, w_out, b_out, (float*)d_out);
}
